// SABlock_88914412961973
// MI455X (gfx1250) — matrix operand order repaired and checked by reading, not compiled, not run on hardware
//
#include <hip/hip_runtime.h>

#define BB 4
#define NN 8192
#define CD 2
#define IND 64
#define OUTD 128
#define KK 16
#define SS 2048              // NN / STRIDE
#define CTOT (IND + CD)      // 66
#define RADIUS_F 0.1f
#define EPS_F 1e-5f
#define S_PER_BLOCK 16

typedef __attribute__((ext_vector_type(16))) _Float16 v16h;
typedef __attribute__((ext_vector_type(8)))  float    v8f;

// ds_swizzle XOR pattern within group-of-32: and=0x1f, or=0, xor=M
#define SWZ_XOR(M) (((M) << 10) | 0x1F)

template <int OFF>
__device__ __forceinline__ float swz_max(float v) {
  return fmaxf(v, __int_as_float(
                      __builtin_amdgcn_ds_swizzle(__float_as_int(v), OFF)));
}

// ---------------- Kernel 1: farthest point sampling (one WG per batch) -----
__global__ __launch_bounds__(1024)
void fps_kernel(const float* __restrict__ xyz, float* __restrict__ sxyz,
                int* __restrict__ sidx) {
  const int b = blockIdx.x;
  const int tid = threadIdx.x;
  const int lane = tid & 31, wid = tid >> 5;
  constexpr int PPT = NN / 1024;  // 8 points per thread, blocked
  __shared__ float s_wv[32];
  __shared__ int   s_wi[32];
  __shared__ int   s_far;
  __shared__ float s_cx, s_cy;

  float px[PPT], py[PPT], dd[PPT];
  const float* xb = xyz + (size_t)b * 2 * NN;
#pragma unroll
  for (int j = 0; j < PPT; ++j) {
    int p = tid * PPT + j;
    px[j] = xb[p];
    py[j] = xb[NN + p];
    dd[j] = 1e10f;
  }
  if (tid == 0) { s_cx = xb[0]; s_cy = xb[NN]; }
  __syncthreads();

  int far = 0;
  for (int i = 0; i < SS; ++i) {
    if (tid == 0) {                     // emit current 'far' (scan semantics)
      sidx[b * SS + i] = far;
      sxyz[(b * 2 + 0) * SS + i] = s_cx;
      sxyz[(b * 2 + 1) * SS + i] = s_cy;
    }
    const float cx = s_cx, cy = s_cy;
    float bv = -1.0f; int bi = 0;
#pragma unroll
    for (int j = 0; j < PPT; ++j) {
      float dx = px[j] - cx, dy = py[j] - cy;
      float d = dx * dx + dy * dy;
      dd[j] = fminf(dd[j], d);
      if (dd[j] > bv) { bv = dd[j]; bi = tid * PPT + j; }
    }
    // wave32 argmax (first-index tiebreak)
#pragma unroll
    for (int off = 16; off >= 1; off >>= 1) {
      float ov = __shfl_xor(bv, off, 32);
      int   oi = __shfl_xor(bi, off, 32);
      if (ov > bv || (ov == bv && oi < bi)) { bv = ov; bi = oi; }
    }
    if (lane == 0) { s_wv[wid] = bv; s_wi[wid] = bi; }
    __syncthreads();
    if (wid == 0) {
      bv = s_wv[lane]; bi = s_wi[lane];
#pragma unroll
      for (int off = 16; off >= 1; off >>= 1) {
        float ov = __shfl_xor(bv, off, 32);
        int   oi = __shfl_xor(bi, off, 32);
        if (ov > bv || (ov == bv && oi < bi)) { bv = ov; bi = oi; }
      }
      if (lane == 0) s_far = bi;
    }
    __syncthreads();
    far = s_far;
    if (far / PPT == tid) { s_cx = px[far % PPT]; s_cy = py[far % PPT]; }
    __syncthreads();
  }
}

// ---------------- Kernel 2: ball query (one wave per sample) ---------------
__global__ __launch_bounds__(256)
void ballq_kernel(const float* __restrict__ xyz, const float* __restrict__ sxyz,
                  int* __restrict__ nidx) {
  const int lane = threadIdx.x & 31;
  const int gw = (blockIdx.x * 256 + threadIdx.x) >> 5;
  if (gw >= BB * SS) return;
  const int b = gw / SS, s = gw % SS;
  const float nx = sxyz[(b * 2 + 0) * SS + s];
  const float ny = sxyz[(b * 2 + 1) * SS + s];
  const float* xb = xyz + (size_t)b * 2 * NN;
  int* np = nidx + ((size_t)b * SS + s) * KK;
  const float r2 = RADIUS_F * RADIUS_F;
  int cnt = 0, first = 0;
  for (int base = 0; base < NN && cnt < KK; base += 32) {
    int j = base + lane;
    float dx = xb[j] - nx, dy = xb[NN + j] - ny;
    unsigned mask = (unsigned)__ballot(dx * dx + dy * dy <= r2);
    while (mask && cnt < KK) {
      int l = __ffs((int)mask) - 1;
      mask &= mask - 1;
      int idx = base + l;
      if (cnt == 0) first = idx;
      if (lane == 0) np[cnt] = idx;
      ++cnt;
    }
  }
  for (int q = cnt; q < KK; ++q)
    if (lane == 0) np[q] = first;
}

// -------- Kernel 3: gather + WMMA GEMM + max-over-neighbors ----------------
// Block: 256 threads = 8 waves; wave w owns output-channel tile w (16 rows).
// Both A (W) and B (feats) are staged in LDS in their WMMA register layouts
// so operand loads are contiguous ds_load_b128 pairs.
struct RunSmem {
  alignas(32) _Float16 feats[2][3 * 32 * 16];  // B operands, 2 samples
  alignas(32) float    dred[8][256];           // per-wave 16x16 f32 D tile
  int   s_idx[2][KK];
  float s_ctr[2][CD];
};

__global__ __launch_bounds__(256)
void gemm_kernel(const float* __restrict__ x, const float* __restrict__ xyz,
                 const float* __restrict__ W, const float* __restrict__ sxyz,
                 const int* __restrict__ nidx, float* __restrict__ h_pre) {
  const int b   = blockIdx.y;
  const int s0  = blockIdx.x * S_PER_BLOCK;
  const int tid = threadIdx.x;
  const int lane = tid & 31, wid = tid >> 5;
  const int half = (lane >= 16) ? 1 : 0;
  const int l16  = lane & 15;

  __shared__ union SMem {
    alignas(32) _Float16 wlds[OUTD * 96];      // A staging (dead after preamble)
    RunSmem r;
  } sm;

  // ---- Preamble: stage W (f32 -> f16, c padded to 96) in A register layout.
  // A layout: lane<16 holds K {0..7,16..23}, lane>=16 holds {8..15,24..31};
  // inverse map for element (row,c): kk=c&31, laneRow=(row&15)+16*((kk>>3)&1),
  // j=(kk&7)+8*((kk>>4)&1).  Packed two consecutive c per dword store.
  {
    unsigned* ww = (unsigned*)sm.wlds;
    for (int p = tid; p < OUTD * 48; p += 256) {
      int row = p / 48;
      int c0  = (p % 48) * 2;
      float v0 = (c0 < CTOT)     ? W[row * CTOT + c0]     : 0.0f;
      float v1 = (c0 + 1 < CTOT) ? W[row * CTOT + c0 + 1] : 0.0f;
      int kk = c0 & 31, st = c0 >> 5, tile = row >> 4;
      int laneRow = (row & 15) + (((kk >> 3) & 1) << 4);
      int j0 = (kk & 7) + (((kk >> 4) & 1) << 3);
      union { _Float16 h[2]; unsigned u; } pk;
      pk.h[0] = (_Float16)v0;
      pk.h[1] = (_Float16)v1;
      ww[(((tile * 3 + st) * 32 + laneRow) * 16 + j0) >> 1] = pk.u;
    }
  }
  __syncthreads();
  v16h a[3];
  {
    const v16h* ap = (const v16h*)sm.wlds;
#pragma unroll
    for (int st = 0; st < 3; ++st) a[st] = ap[(wid * 3 + st) * 32 + lane];
  }
  // loop's first __syncthreads separates wlds reads from RunSmem writes

  const float inv_r = 1.0f / RADIUS_F;
  for (int sl = 0; sl < S_PER_BLOCK; sl += 2) {
    __syncthreads();                       // prev consumers done / wlds dead
    if (tid < 2 * KK) {                    // stage indices + centroids
      int sp = tid >> 4, k = tid & 15;
      sm.r.s_idx[sp][k] = nidx[((size_t)b * SS + (s0 + sl + sp)) * KK + k];
    } else if (tid < 2 * KK + 2 * CD) {
      int t = tid - 2 * KK;
      int sp = t >> 1, d = t & 1;
      sm.r.s_ctr[sp][d] = sxyz[(b * CD + d) * SS + (s0 + sl + sp)];
    }
    __syncthreads();

    // cooperative gather: 768 packed f16-pairs per sample, 3 per thread,
    // written directly in B-operand register layout
    for (int sp = 0; sp < 2; ++sp) {
      unsigned* fw = (unsigned*)sm.r.feats[sp];
      for (int p = tid; p < 768; p += 256) {
        int k   = p & 15;
        int jj  = (p >> 4) & 7;
        int blk = p >> 7;                   // 0..5
        int c0  = blk * 16 + jj * 2;
        int idx = sm.r.s_idx[sp][k];
        float v0 = 0.0f, v1 = 0.0f;
        if (c0 < IND)        v0 = x[((size_t)b * IND + c0) * NN + idx];
        else if (c0 < CTOT)  v0 = (xyz[((size_t)b * CD + (c0 - IND)) * NN + idx]
                                   - sm.r.s_ctr[sp][c0 - IND]) * inv_r;
        int c1 = c0 + 1;
        if (c1 < IND)        v1 = x[((size_t)b * IND + c1) * NN + idx];
        else if (c1 < CTOT)  v1 = (xyz[((size_t)b * CD + (c1 - IND)) * NN + idx]
                                   - sm.r.s_ctr[sp][c1 - IND]) * inv_r;
        union { _Float16 h[2]; unsigned u; } pk;
        pk.h[0] = (_Float16)v0;
        pk.h[1] = (_Float16)v1;
        fw[((blk >> 1) * 32 + k + ((jj >> 2) << 4)) * 8 + (jj & 3) + ((blk & 1) << 2)] = pk.u;
      }
    }
    __syncthreads();

    for (int sp = 0; sp < 2; ++sp) {
      const int s = s0 + sl + sp;
      const v16h* bp = (const v16h*)sm.r.feats[sp];
      v16h b0 = bp[lane];
      v16h b1 = bp[32 + lane];
      v16h b2 = bp[64 + lane];

      v8f acc = {};
      acc = __builtin_amdgcn_wmma_f32_16x16x32_f16(false, a[0], false, b0,
                                                   (short)0, acc, false, false);
      acc = __builtin_amdgcn_wmma_f32_16x16x32_f16(false, a[1], false, b1,
                                                   (short)0, acc, false, false);
      acc = __builtin_amdgcn_wmma_f32_16x16x32_f16(false, a[2], false, b2,
                                                   (short)0, acc, false, false);

      // max over the 16 neighbor columns: transpose D through wave-private
      // LDS (same-wave LDS ops are in-order; no barrier needed), then each
      // lane tree-maxes half a row and one xor-1 swizzle merges halves.
      float* dw = sm.r.dred[wid];
#pragma unroll
      for (int r = 0; r < 8; ++r)
        dw[(r + 8 * half) * 16 + l16] = acc[r];
      asm volatile("s_wait_dscnt 0x0" ::: "memory");
      v8f rowv = *((const v8f*)dw + lane);   // bytes [32*lane, 32*lane+32)
      float t = fmaxf(fmaxf(fmaxf(rowv[0], rowv[1]), fmaxf(rowv[2], rowv[3])),
                      fmaxf(fmaxf(rowv[4], rowv[5]), fmaxf(rowv[6], rowv[7])));
      t = swz_max<SWZ_XOR(1)>(t);
      int o = wid * 16 + (lane >> 1);
      if ((lane & 1) == 0)
        h_pre[((size_t)b * OUTD + o) * SS + s] = t;
    }
  }
}

// ---------------- Kernel 4: BatchNorm over (B,S) + ReLU --------------------
__global__ __launch_bounds__(256)
void bn_kernel(const float* __restrict__ h_pre, const float* __restrict__ gamma,
               const float* __restrict__ beta, float* __restrict__ h_out) {
  const int o = blockIdx.x;
  const int tid = threadIdx.x;
  __shared__ float s_sum[256], s_sq[256];
  float sum = 0.0f, sq = 0.0f;
  for (int i = tid; i < BB * SS; i += 256) {
    int b = i / SS, s = i % SS;
    float v = h_pre[((size_t)b * OUTD + o) * SS + s];
    sum += v; sq += v * v;
  }
  s_sum[tid] = sum; s_sq[tid] = sq;
  __syncthreads();
  for (int off = 128; off >= 1; off >>= 1) {
    if (tid < off) { s_sum[tid] += s_sum[tid + off]; s_sq[tid] += s_sq[tid + off]; }
    __syncthreads();
  }
  const float inv_n = 1.0f / (float)(BB * SS);
  float mean = s_sum[0] * inv_n;
  float var  = s_sq[0] * inv_n - mean * mean;
  float scale = gamma[o] * rsqrtf(var + EPS_F);
  float shift = beta[o] - mean * scale;
  for (int i = tid; i < BB * SS; i += 256) {
    int b = i / SS, s = i % SS;
    float v = h_pre[((size_t)b * OUTD + o) * SS + s];
    h_out[((size_t)b * OUTD + o) * SS + s] = fmaxf(v * scale + shift, 0.0f);
  }
}

// ---------------------------------------------------------------------------
extern "C" void kernel_launch(void* const* d_in, const int* in_sizes, int n_in,
                              void* d_out, int out_size, void* d_ws, size_t ws_size,
                              hipStream_t stream) {
  const float* x     = (const float*)d_in[0];
  const float* xyz   = (const float*)d_in[1];
  const float* W     = (const float*)d_in[2];
  const float* gamma = (const float*)d_in[3];
  const float* beta  = (const float*)d_in[4];

  // Output tuple concatenated flat: h, sample_xyz, sample_idx, nidx
  float* out   = (float*)d_out;
  float* h_out = out;                                    // B*128*S f32
  float* sxyz  = out + (size_t)BB * OUTD * SS;           // B*2*S   f32
  int*   sidx  = (int*)(sxyz + (size_t)BB * CD * SS);    // B*S     i32
  int*   nidx  = (int*)(sidx + (size_t)BB * SS);         // B*S*K   i32

  float* h_pre = (float*)d_ws;                           // B*128*S f32 scratch

  fps_kernel<<<BB, 1024, 0, stream>>>(xyz, sxyz, sidx);
  ballq_kernel<<<(BB * SS * 32) / 256, 256, 0, stream>>>(xyz, sxyz, nidx);
  dim3 g(SS / S_PER_BLOCK, BB);
  gemm_kernel<<<g, 256, 0, stream>>>(x, xyz, W, sxyz, nidx, h_pre);
  bn_kernel<<<OUTD, 256, 0, stream>>>(h_pre, gamma, beta, h_out);
}
